// AdvectionNet_2791728742925
// MI455X (gfx1250) — matrix operand order repaired and checked by reading, not compiled, not run on hardware
//
#include <hip/hip_runtime.h>

typedef __attribute__((ext_vector_type(16))) _Float16 v16h;
typedef __attribute__((ext_vector_type(8)))  _Float16 v8h;
typedef __attribute__((ext_vector_type(8)))  float    v8f;
typedef __attribute__((ext_vector_type(4)))  unsigned int v4u;
typedef __attribute__((ext_vector_type(8)))  unsigned int v8u;

#define DEVI __device__ __forceinline__

DEVI int iclampd(int v, int lo, int hi) { return v < lo ? lo : (v > hi ? hi : v); }

// ---------------------------------------------------------------------------
// Workspace layout (bytes). All chunk sizes are multiples of 256.
// ---------------------------------------------------------------------------
static constexpr size_t MT = 65536;            // B*N points
static constexpr size_t OFF_Y    = 0;                        // f32 GEMM out, up to M x 1024
static constexpr size_t OFF_H    = OFF_Y    + MT * 1024 * 4; // f16, M x 1344 (reused for f2..f4 acts)
static constexpr size_t OFF_VAL  = OFF_H    + MT * 1344 * 2; // f16, M x 320 (concatenated value)
static constexpr size_t OFF_A0   = OFF_VAL  + MT * 320 * 2;  // f16, M x 64 (45 used)
static constexpr size_t OFF_AA   = OFF_A0   + MT * 64 * 2;   // f16, M x 64
static constexpr size_t OFF_AB   = OFF_AA   + MT * 64 * 2;   // f16, M x 64
static constexpr size_t OFF_VEL  = OFF_AB   + MT * 64 * 2;   // f32, M x 3
static constexpr size_t OFF_P    = OFF_VEL  + MT * 3 * 4;    // f32, M x 3
static constexpr size_t OFF_STAT = OFF_P    + MT * 3 * 4;    // f32, 2*1024
static constexpr size_t OFF_GMAX = OFF_STAT + 2048 * 4;      // f32, 8*1024
static constexpr size_t OFF_CELL = OFF_GMAX + 8192 * 4;      // f32, up to 8*12^3 cells * 4
static constexpr size_t OFF_FNUM = OFF_CELL + 8 * 1728 * 4 * 4; // f32, 1728 vox * 32
static constexpr size_t OFF_FDEN = OFF_FNUM + 1728 * 32 * 4;    // f32, 1728
static constexpr size_t OFF_F16A = OFF_FDEN + 1728 * 4;         // f16, 1728 x 32
static constexpr size_t OFF_F16B = OFF_F16A + 1728 * 32 * 2;    // f16, 1728 x 32
static constexpr size_t OFF_F16C = OFF_F16B + 1728 * 32 * 2;    // f16, 1728 x 32
static constexpr size_t OFF_FY   = OFF_F16C + 1728 * 32 * 2;    // f32, 1728 x 32
static constexpr size_t OFF_FST  = OFF_FY   + 1728 * 32 * 4;    // f32, 2*32
static constexpr size_t OFF_NVEL = OFF_FST  + 256;              // f32, 1728 x 3
static constexpr size_t OFF_VG   = OFF_NVEL + 1728 * 3 * 4;     // f32, 1728 x 3
static constexpr size_t OFF_PENS = OFF_VG   + 1728 * 3 * 4;     // f32, 240 used
static constexpr size_t OFF_W16  = OFF_PENS + 1024;             // f16 weights (padded/transposed), ~2.7 MB

// ---------------------------------------------------------------------------
// Tensor Data Mover descriptor helpers (CDNA5 ISA ch.8: D# groups 0/1, 2D tile)
// ---------------------------------------------------------------------------
DEVI v4u tdm_group0(unsigned lds_addr, unsigned long long gaddr)
{
    v4u g;
    g.x = 1u;                                    // count=1 (valid), user mode, no gather
    g.y = lds_addr;                              // bits 63:32 : LDS byte address
    g.z = (unsigned)(gaddr & 0xFFFFFFFFu);       // bits 95:64 : global addr lo
    g.w = (unsigned)((gaddr >> 32) & 0x01FFFFFFu) | (2u << 30);  // addr hi | type=2
    return g;
}
// data_size: 0=1B,1=2B,2=4B,3=8B. pad_interval: 2^(v+1) DWORDs. pad_amount: v+1 DWORDs.
DEVI v8u tdm_group1(unsigned data_size, unsigned pad_en, unsigned pad_interval,
                    unsigned pad_amount, unsigned td0, unsigned td1,
                    unsigned tile0, unsigned tile1, unsigned long long stride0)
{
    v8u g;
    g[0] = (data_size << 16) | (pad_en << 20) | (pad_interval << 22) | (pad_amount << 25);
    g[1] = (td0 & 0xFFFFu) << 16;                         // tensor_dim0 lo16 (bits 63:48)
    g[2] = (td0 >> 16) | ((td1 & 0xFFFFu) << 16);         // tensor_dim0 hi16 | tensor_dim1 lo16
    g[3] = ((td1 >> 16) & 0xFFFFu) | ((tile0 & 0xFFFFu) << 16);  // tensor_dim1 hi16 | tile_dim0
    g[4] = (tile1 & 0xFFFFu);                             // tile_dim1 | tile_dim2=0
    g[5] = (unsigned)(stride0 & 0xFFFFFFFFu);             // tensor_dim0_stride lo32
    g[6] = (unsigned)((stride0 >> 32) & 0xFFFFu);         // stride hi16 | dim1_stride lo16=0
    g[7] = 0u;
    return g;
}
DEVI void tdm_load(v4u g0, v8u g1)
{
    // 2D tensor: only descriptor groups 0 and 1 (VADDR2/3 = NULL form)
    asm volatile("tensor_load_to_lds %0, %1" :: "s"(g0), "s"(g1) : "memory");
}

// ---------------------------------------------------------------------------
// f16 WMMA GEMM:  C[M,N] (f32) = A[M,Kp] (f16 row-major, lda, zero-padded)
//                              * WT[Kp,Np] (f16, pre-transposed+padded) (+bias)
// Tile staging via Tensor Data Mover (wave 0 issues descriptors; TDM zero-fills
// rows past M and applies the LDS padding strides). Double-buffered, one
// barrier per K-step; other 7 waves spend issue slots only on ds_load + WMMA.
// ---------------------------------------------------------------------------
__launch_bounds__(256, 2)
__global__ void gemm_wmma_kernel(const _Float16* __restrict__ A, int lda,
                                 const _Float16* __restrict__ WT, int ldw,
                                 const float* __restrict__ bias,
                                 float* __restrict__ C, int ldc,
                                 int M, int N, int Kloop)
{
    constexpr int BM = 128, BN = 64, BK = 32;
    constexpr int ASTR = 56;   // halves; 112B row stride = 64B row + 48B pad (TDM pad 3/11)
    constexpr int BSTR = 72;   // halves; 144B row stride = 128B row + 16B pad (TDM pad 4/3)
    constexpr int ATILE = BM * ASTR, BTILE = BK * BSTR;
    __shared__ _Float16 sA[2 * ATILE];
    __shared__ _Float16 sB[2 * BTILE];

    const int tid  = threadIdx.x;
    const int wave = tid >> 5;
    const int lane = tid & 31;
    const int m0 = blockIdx.x * BM;
    const int n0 = blockIdx.y * BN;

    const unsigned rowsA = (unsigned)(M - m0);   // TDM returns zeros for rows >= M
    const unsigned ldsA0 = (unsigned)(size_t)(&sA[0]);
    const unsigned ldsB0 = (unsigned)(size_t)(&sB[0]);

    auto issue_tiles = [&](int k0, int buf) {
        // A tile: 128 rows x 32 halves (64B), global row stride lda halves
        v4u a0 = tdm_group0(ldsA0 + (unsigned)(buf * ATILE * 2),
                            (unsigned long long)(size_t)A + ((size_t)m0 * lda + k0) * 2);
        v8u a1 = tdm_group1(1u, 1u, 3u, 11u, 32u, rowsA, 32u, 128u,
                            (unsigned long long)(unsigned)lda);
        tdm_load(a0, a1);
        // B tile: 32 rows x 64 halves (128B), global row stride ldw halves
        v4u b0 = tdm_group0(ldsB0 + (unsigned)(buf * BTILE * 2),
                            (unsigned long long)(size_t)WT + ((size_t)k0 * ldw + n0) * 2);
        v8u b1 = tdm_group1(1u, 1u, 4u, 3u, 64u, (unsigned)(Kloop - k0), 64u, 32u,
                            (unsigned long long)(unsigned)ldw);
        tdm_load(b0, b1);
    };

    const int laneR = lane & 15;
    const int kbase = (lane < 16) ? 0 : 8;   // 16-bit A layout: K 0-7/16-23 vs 8-15/24-31

    v8f acc0 = {}, acc1 = {}, acc2 = {}, acc3 = {};

    if (wave == 0) {
        issue_tiles(0, 0);
        __builtin_amdgcn_s_wait_tensorcnt(0);
    }
    __syncthreads();

    int cur = 0;
    for (int k0 = 0; k0 < Kloop; k0 += BK) {
        const bool more = (k0 + BK) < Kloop;
        if (more && wave == 0) issue_tiles(k0 + BK, cur ^ 1);  // DMA overlaps WMMA below

        union { v16h v; v8h h[2]; } af, bf0, bf1, bf2, bf3;
        const _Float16* ap = sA + cur * ATILE + (wave * 16 + laneR) * ASTR + kbase;
        af.h[0] = *(const v8h*)(ap);
        af.h[1] = *(const v8h*)(ap + 16);
        const _Float16* bp = sB + cur * BTILE + kbase * BSTR + laneR;  // lane&15 == n, elements over k (table order)
#pragma unroll
        for (int s = 0; s < 8; ++s) {
            bf0.v[s] = bp[s * BSTR + 0];  bf0.v[8 + s] = bp[(16 + s) * BSTR + 0];
            bf1.v[s] = bp[s * BSTR + 16]; bf1.v[8 + s] = bp[(16 + s) * BSTR + 16];
            bf2.v[s] = bp[s * BSTR + 32]; bf2.v[8 + s] = bp[(16 + s) * BSTR + 32];
            bf3.v[s] = bp[s * BSTR + 48]; bf3.v[8 + s] = bp[(16 + s) * BSTR + 48];
        }

        acc0 = __builtin_amdgcn_wmma_f32_16x16x32_f16(false, af.v, false, bf0.v, (short)0, acc0, false, false);
        acc1 = __builtin_amdgcn_wmma_f32_16x16x32_f16(false, af.v, false, bf1.v, (short)0, acc1, false, false);
        acc2 = __builtin_amdgcn_wmma_f32_16x16x32_f16(false, af.v, false, bf2.v, (short)0, acc2, false, false);
        acc3 = __builtin_amdgcn_wmma_f32_16x16x32_f16(false, af.v, false, bf3.v, (short)0, acc3, false, false);

        if (wave == 0) __builtin_amdgcn_s_wait_tensorcnt(0);  // next tiles landed
        __syncthreads();
        cur ^= 1;
    }

    // D layout: VGPR v -> row v (lanes 0-15) / v+8 (lanes 16-31); lane&15 -> col
    const int rbase = m0 + wave * 16 + (lane >> 4) * 8;
    const int cl = lane & 15;
    v8f accs[4] = {acc0, acc1, acc2, acc3};
#pragma unroll
    for (int nt = 0; nt < 4; ++nt) {
        int col = n0 + nt * 16 + cl;
        if (col < N) {
            float badd = bias ? bias[col] : 0.f;
#pragma unroll
            for (int v = 0; v < 8; ++v) {
                int row = rbase + v;
                if (row < M) C[(size_t)row * ldc + col] = accs[nt][v] + badd;
            }
        }
    }
}

// ---------------------------------------------------------------------------
// Utility kernels
// ---------------------------------------------------------------------------
__global__ void fill_f32_kernel(float* p, float v, int n) {
    int i = blockIdx.x * 256 + threadIdx.x;
    if (i < n) p[i] = v;
}
// weight (N x K f32, row-major) -> transposed, padded f16 (Kp x Np), zero-filled
__global__ void cvt_wt_kernel(const float* __restrict__ src, _Float16* __restrict__ dst,
                              int N, int K, int Np, int Kp)
{
    int i = blockIdx.x * 256 + threadIdx.x;
    if (i >= Kp * Np) return;
    int k = i / Np, n = i - k * Np;
    dst[i] = (k < K && n < N) ? (_Float16)src[(size_t)n * K + k] : (_Float16)0.f;
}

// BatchNorm: stats (sum, sumsq) with atomics, then apply+ReLU -> f16 at column offset
__global__ void bn_stats_kernel(const float* __restrict__ Y, float* __restrict__ stat,
                                int M, int C, int rowsPerBlk)
{
    int c = blockIdx.x * 256 + threadIdx.x;
    if (c >= C) return;
    int r0 = blockIdx.y * rowsPerBlk;
    int r1 = r0 + rowsPerBlk; if (r1 > M) r1 = M;
    float s = 0.f, s2 = 0.f;
    for (int r = r0; r < r1; ++r) {
        float v = Y[(size_t)r * C + c];
        s += v; s2 += v * v;
    }
    atomicAdd(&stat[c], s);
    atomicAdd(&stat[C + c], s2);
}
__global__ void bn_apply_kernel(const float* __restrict__ Y, const float* __restrict__ stat,
                                const float* __restrict__ g, const float* __restrict__ be,
                                _Float16* __restrict__ out, int M, int C, int ldo, int cofs,
                                float invM)
{
    int i = blockIdx.x * 256 + threadIdx.x;
    if (i >= M * C) return;
    int r = i / C, c = i - r * C;
    float mean = stat[c] * invM;
    float var  = stat[C + c] * invM - mean * mean;
    float v = (Y[(size_t)r * C + c] - mean) * rsqrtf(var + 1e-5f) * g[c] + be[c];
    out[(size_t)r * ldo + cofs + c] = (_Float16)fmaxf(v, 0.f);
}
// Fused BN + ReLU + per-batch max (for gmax over ic2 output)
__global__ void bn_max_kernel(const float* __restrict__ Y, const float* __restrict__ stat,
                              const float* __restrict__ g, const float* __restrict__ be,
                              float* __restrict__ gmax, int Npts, int C, float invM)
{
    int c = blockIdx.x * 256 + threadIdx.x;
    int b = blockIdx.y;
    if (c >= C) return;
    float mean = stat[c] * invM;
    float var  = stat[C + c] * invM - mean * mean;
    float sc = rsqrtf(var + 1e-5f) * g[c];
    float sh = be[c] - mean * sc;
    float mx = -3.4e38f;
    const float* base = Y + (size_t)b * Npts * C + c;
    for (int n = 0; n < Npts; ++n) {
        float v = base[(size_t)n * C] * sc + sh;
        v = fmaxf(v, 0.f);
        mx = fmaxf(mx, v);
    }
    gmax[b * C + c] = mx;
}

// value0: xyz into cols 0..2, zero pad cols 45..63
__global__ void init_act0_kernel(const float* __restrict__ pc, _Float16* __restrict__ a0, int Mtot)
{
    int i = blockIdx.x * 256 + threadIdx.x;
    if (i >= Mtot) return;
    _Float16* r = a0 + (size_t)i * 64;
    r[0] = (_Float16)pc[i * 3 + 0];
    r[1] = (_Float16)pc[i * 3 + 1];
    r[2] = (_Float16)pc[i * 3 + 2];
#pragma unroll
    for (int c = 45; c < 64; ++c) r[c] = (_Float16)0.f;
}

// grid_avg: scatter centroid sums, then gather mean + offset (6 channels)
__global__ void grid_scatter_kernel(const float* __restrict__ pc, float* __restrict__ cell,
                                    int gs, int Mtot, int Npts)
{
    int i = blockIdx.x * 256 + threadIdx.x;
    if (i >= Mtot) return;
    int b = i / Npts;
    float x = pc[i * 3], y = pc[i * 3 + 1], z = pc[i * 3 + 2];
    int cx = iclampd((int)floorf((x + 1.f) * 0.5f * gs), 0, gs - 1);
    int cy = iclampd((int)floorf((y + 1.f) * 0.5f * gs), 0, gs - 1);
    int cz = iclampd((int)floorf((z + 1.f) * 0.5f * gs), 0, gs - 1);
    int cid = ((b * gs + cx) * gs + cy) * gs + cz;
    atomicAdd(&cell[cid * 4 + 0], x);
    atomicAdd(&cell[cid * 4 + 1], y);
    atomicAdd(&cell[cid * 4 + 2], z);
    atomicAdd(&cell[cid * 4 + 3], 1.f);
}
__global__ void grid_gather_kernel(const float* __restrict__ pc, const float* __restrict__ cell,
                                   _Float16* __restrict__ a0, int cofs, int gs, int Mtot, int Npts)
{
    int i = blockIdx.x * 256 + threadIdx.x;
    if (i >= Mtot) return;
    int b = i / Npts;
    float x = pc[i * 3], y = pc[i * 3 + 1], z = pc[i * 3 + 2];
    int cx = iclampd((int)floorf((x + 1.f) * 0.5f * gs), 0, gs - 1);
    int cy = iclampd((int)floorf((y + 1.f) * 0.5f * gs), 0, gs - 1);
    int cz = iclampd((int)floorf((z + 1.f) * 0.5f * gs), 0, gs - 1);
    int cid = ((b * gs + cx) * gs + cy) * gs + cz;
    float inv = 1.f / fmaxf(cell[cid * 4 + 3], 1.f);
    float mx = cell[cid * 4 + 0] * inv, my = cell[cid * 4 + 1] * inv, mz = cell[cid * 4 + 2] * inv;
    _Float16* r = a0 + (size_t)i * 64 + cofs;
    r[0] = (_Float16)mx; r[1] = (_Float16)my; r[2] = (_Float16)mz;
    r[3] = (_Float16)(x - mx); r[4] = (_Float16)(y - my); r[5] = (_Float16)(z - mz);
}

__global__ void add3_kernel(const float* __restrict__ a, const float* __restrict__ b,
                            float* __restrict__ o, int n)
{
    int i = blockIdx.x * 256 + threadIdx.x;
    if (i < n) o[i] = a[i] + b[i];
}

// RevTrilinear scatter: point features (f16 or f32) -> voxel grid num/den
__global__ void splat_scatter_kernel(const float* __restrict__ p,
                                     const _Float16* __restrict__ v16, const float* __restrict__ v32,
                                     int ldv, int cofs, int C,
                                     float* __restrict__ num, float* __restrict__ den,
                                     int Mtot, int Npts, int gs)
{
    int i = blockIdx.x * 256 + threadIdx.x;
    if (i >= Mtot) return;
    int b = i / Npts;
    float ux = (p[i * 3 + 0] + 1.f) * 0.5f * (gs - 1);
    float uy = (p[i * 3 + 1] + 1.f) * 0.5f * (gs - 1);
    float uz = (p[i * 3 + 2] + 1.f) * 0.5f * (gs - 1);
    int ix = (int)floorf(ux), iy = (int)floorf(uy), iz = (int)floorf(uz);
    float fx = ux - ix, fy = uy - iy, fz = uz - iz;
    float vals[32];
    for (int ch = 0; ch < C; ++ch)
        vals[ch] = v16 ? (float)v16[(size_t)i * ldv + cofs + ch] : v32[(size_t)i * ldv + cofs + ch];
#pragma unroll
    for (int c = 0; c < 8; ++c) {
        int cx = ix + ((c >> 2) & 1), cy = iy + ((c >> 1) & 1), cz = iz + (c & 1);
        if (cx < 0 || cx >= gs || cy < 0 || cy >= gs || cz < 0 || cz >= gs) continue;
        float w = ((c & 4) ? fx : 1.f - fx) * ((c & 2) ? fy : 1.f - fy) * ((c & 1) ? fz : 1.f - fz);
        int cid = ((b * gs + cx) * gs + cy) * gs + cz;
        if (den) atomicAdd(&den[cid], w);
        float* np = num + (size_t)cid * C;
        for (int ch = 0; ch < C; ++ch) atomicAdd(&np[ch], w * vals[ch]);
    }
}
__global__ void field_div_f16_kernel(const float* __restrict__ num, const float* __restrict__ den,
                                     _Float16* __restrict__ out, int V, int C)
{
    int i = blockIdx.x * 256 + threadIdx.x;
    if (i >= V * C) return;
    out[i] = (_Float16)(num[i] / (den[i / C] + 1e-8f));
}
__global__ void field_div_f32_kernel(const float* __restrict__ num, const float* __restrict__ den,
                                     float* __restrict__ out, int V, int C)
{
    int i = blockIdx.x * 256 + threadIdx.x;
    if (i >= V * C) return;
    out[i] = num[i] / (den[i / C] + 1e-8f);
}

// direct 3x3x3 conv on the 6^3 grid, voxel-major (b,x,y,z) x C
__global__ void conv3d_k3_kernel(const _Float16* __restrict__ in, const float* __restrict__ w,
                                 const float* __restrict__ bias, float* __restrict__ out,
                                 int Bn, int gs, int C)
{
    int idx = blockIdx.x * 256 + threadIdx.x;
    int total = Bn * gs * gs * gs * C;
    if (idx >= total) return;
    int oc = idx % C;
    int vox = idx / C;
    int z = vox % gs, y = (vox / gs) % gs, x = (vox / (gs * gs)) % gs, b = vox / (gs * gs * gs);
    float acc = bias[oc];
    for (int dx = -1; dx <= 1; ++dx) {
        int xx = x + dx; if (xx < 0 || xx >= gs) continue;
        for (int dy = -1; dy <= 1; ++dy) {
            int yy = y + dy; if (yy < 0 || yy >= gs) continue;
            for (int dz = -1; dz <= 1; ++dz) {
                int zz = z + dz; if (zz < 0 || zz >= gs) continue;
                const _Float16* ip = in + (size_t)(((b * gs + xx) * gs + yy) * gs + zz) * C;
                int t = (dx + 1) * 9 + (dy + 1) * 3 + (dz + 1);
                for (int ic = 0; ic < C; ++ic)
                    acc += (float)ip[ic] * w[(size_t)(oc * C + ic) * 27 + t];
            }
        }
    }
    out[(size_t)vox * C + oc] = acc;
}

// new_vel = gvw @ field + gvb  (per voxel, 3 out channels)
__global__ void newvel_kernel(const _Float16* __restrict__ field, const float* __restrict__ gvw,
                              const float* __restrict__ gvb, float* __restrict__ nv, int V, int C)
{
    int v = blockIdx.x * 256 + threadIdx.x;
    if (v >= V) return;
    for (int o = 0; o < 3; ++o) {
        float acc = gvb[o];
        for (int ic = 0; ic < C; ++ic) acc += (float)field[(size_t)v * C + ic] * gvw[o * C + ic];
        nv[v * 3 + o] = acc;
    }
}

// trilinear gather of 32-ch grid at points -> f16 columns of value
__global__ void sample_feature_kernel(const _Float16* __restrict__ grid, const float* __restrict__ p,
                                      _Float16* __restrict__ val, int ldv, int cofs,
                                      int Mtot, int Npts, int gs)
{
    int i = blockIdx.x * 256 + threadIdx.x;
    if (i >= Mtot) return;
    int b = i / Npts;
    float ux = (p[i * 3 + 0] + 1.f) * 0.5f * (gs - 1);
    float uy = (p[i * 3 + 1] + 1.f) * 0.5f * (gs - 1);
    float uz = (p[i * 3 + 2] + 1.f) * 0.5f * (gs - 1);
    int ix = (int)floorf(ux), iy = (int)floorf(uy), iz = (int)floorf(uz);
    float fx = ux - ix, fy = uy - iy, fz = uz - iz;
    float acc[32];
    for (int ch = 0; ch < 32; ++ch) acc[ch] = 0.f;
#pragma unroll
    for (int c = 0; c < 8; ++c) {
        int cx = ix + ((c >> 2) & 1), cy = iy + ((c >> 1) & 1), cz = iz + (c & 1);
        if (cx < 0 || cx >= gs || cy < 0 || cy >= gs || cz < 0 || cz >= gs) continue;
        float w = ((c & 4) ? fx : 1.f - fx) * ((c & 2) ? fy : 1.f - fy) * ((c & 1) ? fz : 1.f - fz);
        const _Float16* gp = grid + (size_t)(((b * gs + cx) * gs + cy) * gs + cz) * 32;
        for (int ch = 0; ch < 32; ++ch) acc[ch] += w * (float)gp[ch];
    }
    _Float16* r = val + (size_t)i * ldv + cofs;
    for (int ch = 0; ch < 32; ++ch) r[ch] = (_Float16)acc[ch];
}

// PIC/FLIP velocity update + advect positions
__global__ void flip_update_kernel(float* __restrict__ p, float* __restrict__ vel,
                                   const float* __restrict__ nv, const float* __restrict__ vg,
                                   int Mtot, int Npts, int gs)
{
    int i = blockIdx.x * 256 + threadIdx.x;
    if (i >= Mtot) return;
    int b = i / Npts;
    float ux = (p[i * 3 + 0] + 1.f) * 0.5f * (gs - 1);
    float uy = (p[i * 3 + 1] + 1.f) * 0.5f * (gs - 1);
    float uz = (p[i * 3 + 2] + 1.f) * 0.5f * (gs - 1);
    int ix = (int)floorf(ux), iy = (int)floorf(uy), iz = (int)floorf(uz);
    float fx = ux - ix, fy = uy - iy, fz = uz - iz;
    float sn[3] = {0.f, 0.f, 0.f}, sg[3] = {0.f, 0.f, 0.f};
#pragma unroll
    for (int c = 0; c < 8; ++c) {
        int cx = ix + ((c >> 2) & 1), cy = iy + ((c >> 1) & 1), cz = iz + (c & 1);
        if (cx < 0 || cx >= gs || cy < 0 || cy >= gs || cz < 0 || cz >= gs) continue;
        float w = ((c & 4) ? fx : 1.f - fx) * ((c & 2) ? fy : 1.f - fy) * ((c & 1) ? fz : 1.f - fz);
        int cid = ((b * gs + cx) * gs + cy) * gs + cz;
        for (int d = 0; d < 3; ++d) {
            sn[d] += w * nv[cid * 3 + d];
            sg[d] += w * vg[cid * 3 + d];
        }
    }
    float invG = 1.f / (float)gs;
    for (int d = 0; d < 3; ++d) {
        float pic  = sn[d] * invG;
        float flip = (sn[d] - sg[d]) * invG;
        float v = (vel[i * 3 + d] + flip) * 0.3f + pic * 0.7f;
        vel[i * 3 + d] = v;
        p[i * 3 + d] += v;
    }
}

// h = concat(value[320], gmax[1024] broadcast per batch)
__global__ void build_h_kernel(const _Float16* __restrict__ val, const float* __restrict__ gmax,
                               _Float16* __restrict__ h, int Mtot, int Npts)
{
    int i = blockIdx.x * 256 + threadIdx.x;
    if (i >= Mtot * 1344) return;
    int r = i / 1344, c = i - r * 1344;
    h[i] = (c < 320) ? val[(size_t)r * 320 + c]
                     : (_Float16)gmax[(r / Npts) * 1024 + (c - 320)];
}

__global__ void log_softmax_kernel(const float* __restrict__ Y, float* __restrict__ out,
                                   int Mtot, int C)
{
    int r = blockIdx.x * 256 + threadIdx.x;
    if (r >= Mtot) return;
    const float* row = Y + (size_t)r * C;
    float mx = -3.4e38f;
    for (int c = 0; c < C; ++c) mx = fmaxf(mx, row[c]);
    float s = 0.f;
    for (int c = 0; c < C; ++c) s += expf(row[c] - mx);
    float lg = logf(s) + mx;
    float* orow = out + (size_t)r * C;
    for (int c = 0; c < C; ++c) orow[c] = row[c] - lg;
}

// penalty: pens[0..47]=cnt, [48..191]=sum3, [192..239]=dsum
__global__ void pen_pass1_kernel(const float* __restrict__ p, const int* __restrict__ seg,
                                 float* __restrict__ pens, int Mtot, int Npts)
{
    int i = blockIdx.x * 256 + threadIdx.x;
    if (i >= Mtot) return;
    int s = (i / Npts) * 6 + seg[i];
    atomicAdd(&pens[s], 1.f);
    atomicAdd(&pens[48 + s * 3 + 0], p[i * 3 + 0]);
    atomicAdd(&pens[48 + s * 3 + 1], p[i * 3 + 1]);
    atomicAdd(&pens[48 + s * 3 + 2], p[i * 3 + 2]);
}
__global__ void pen_pass2_kernel(const float* __restrict__ p, const int* __restrict__ seg,
                                 float* __restrict__ pens, int Mtot, int Npts)
{
    int i = blockIdx.x * 256 + threadIdx.x;
    if (i >= Mtot) return;
    int s = (i / Npts) * 6 + seg[i];
    float inv = 1.f / fmaxf(pens[s], 1.f);
    float dx = p[i * 3 + 0] - pens[48 + s * 3 + 0] * inv;
    float dy = p[i * 3 + 1] - pens[48 + s * 3 + 1] * inv;
    float dz = p[i * 3 + 2] - pens[48 + s * 3 + 2] * inv;
    atomicAdd(&pens[192 + s], sqrtf(dx * dx + dy * dy + dz * dz + 1e-12f));
}
__global__ void pen_final_kernel(const float* __restrict__ pens, float* __restrict__ out)
{
    if (threadIdx.x != 0 || blockIdx.x != 0) return;
    float t = 0.f;
    for (int s = 0; s < 48; ++s) {
        float cnt = pens[s];
        if (cnt <= 0.f) continue;
        float inv = 1.f / fmaxf(cnt, 1.f);
        float cx = pens[48 + s * 3 + 0] * inv;
        float cy = pens[48 + s * 3 + 1] * inv;
        float cz = pens[48 + s * 3 + 2] * inv;
        t += fmaxf(pens[192 + s] * inv, 0.1f);
        t += fmaxf(sqrtf(cx * cx + cy * cy + cz * cz + 1e-12f) - 1.f, 0.f);
    }
    for (int b = 0; b < 8; ++b)
        for (int l1 = 0; l1 < 6; ++l1)
            for (int l2 = l1 + 1; l2 < 6; ++l2) {
                int s1 = b * 6 + l1, s2 = b * 6 + l2;
                if (pens[s1] <= 0.f || pens[s2] <= 0.f) continue;
                float i1 = 1.f / fmaxf(pens[s1], 1.f), i2 = 1.f / fmaxf(pens[s2], 1.f);
                float dx = pens[48 + s1 * 3 + 0] * i1 - pens[48 + s2 * 3 + 0] * i2;
                float dy = pens[48 + s1 * 3 + 1] * i1 - pens[48 + s2 * 3 + 1] * i2;
                float dz = pens[48 + s1 * 3 + 2] * i1 - pens[48 + s2 * 3 + 2] * i2;
                float pd = sqrtf(dx * dx + dy * dy + dz * dz + 1e-12f);
                t += fmaxf(1.5f - pd, 0.f);
            }
    out[0] = t * 0.125f;
}

// ---------------------------------------------------------------------------
// Host orchestration
// ---------------------------------------------------------------------------
extern "C" void kernel_launch(void* const* d_in, const int* in_sizes, int n_in,
                              void* d_out, int out_size, void* d_ws, size_t ws_size,
                              hipStream_t stream)
{
    (void)in_sizes; (void)n_in; (void)out_size; (void)ws_size;
    const int Mtot = 65536, Npts = 8192, Bv = 8 * 216, G = 6;
    static const int GRIDS[7] = {12, 10, 8, 6, 4, 3, 2};

    // ---- unpack inputs (JAX pytree order: sorted dict keys, lists in order) ----
    int cur = 0;
    const float* pc  = (const float*)d_in[cur++];
    const int*   seg = (const int*)d_in[cur++];
    struct Blk { const float *b, *be, *g, *w; };
    auto blk = [&]() -> Blk {
        Blk r;
        r.b  = (const float*)d_in[cur++];
        r.be = (const float*)d_in[cur++];
        r.g  = (const float*)d_in[cur++];
        r.w  = (const float*)d_in[cur++];
        return r;
    };
    Blk c1b[4], c2b[4], fcb[4], gfb[4];
    const float *gvb[4], *gvw[4];
    for (int i = 0; i < 4; ++i) {
        c1b[i] = blk(); c2b[i] = blk(); fcb[i] = blk(); gfb[i] = blk();
        gvb[i] = (const float*)d_in[cur++];
        gvw[i] = (const float*)d_in[cur++];
    }
    Blk f1 = blk(), f2 = blk(), f3 = blk(), f4 = blk();
    const float* fb = (const float*)d_in[cur++];
    const float* fw = (const float*)d_in[cur++];
    Blk ic1a = blk(), ic1b = blk(), ic2 = blk(), iva = blk();
    const float* ivb = (const float*)d_in[cur++];
    const float* ivw = (const float*)d_in[cur++];

    // ---- workspace ----
    char* ws = (char*)d_ws;
    float*    Y    = (float*)(ws + OFF_Y);
    _Float16* H    = (_Float16*)(ws + OFF_H);
    _Float16* VAL  = (_Float16*)(ws + OFF_VAL);
    _Float16* A0   = (_Float16*)(ws + OFF_A0);
    _Float16* AA   = (_Float16*)(ws + OFF_AA);
    _Float16* AB   = (_Float16*)(ws + OFF_AB);
    float*    VEL  = (float*)(ws + OFF_VEL);
    float*    P    = (float*)(ws + OFF_P);
    float*    STAT = (float*)(ws + OFF_STAT);
    float*    GMAX = (float*)(ws + OFF_GMAX);
    float*    CELL = (float*)(ws + OFF_CELL);
    float*    FNUM = (float*)(ws + OFF_FNUM);
    float*    FDEN = (float*)(ws + OFF_FDEN);
    _Float16* F16A = (_Float16*)(ws + OFF_F16A);
    _Float16* F16B = (_Float16*)(ws + OFF_F16B);
    _Float16* F16C = (_Float16*)(ws + OFF_F16C);
    float*    FY   = (float*)(ws + OFF_FY);
    float*    FST  = (float*)(ws + OFF_FST);
    float*    NVEL = (float*)(ws + OFF_NVEL);
    float*    VG   = (float*)(ws + OFF_VG);
    float*    PENS = (float*)(ws + OFF_PENS);
    _Float16* W16  = (_Float16*)(ws + OFF_W16);

    auto nb = [](int n) { return (n + 255) / 256; };
    auto fill = [&](float* p_, float v, int n) {
        fill_f32_kernel<<<nb(n), 256, 0, stream>>>(p_, v, n);
    };

    struct WtT { _Float16* p; int ldw; int kp; };
    size_t wo = 0;
    auto wcvt = [&](const float* src, int N, int K) -> WtT {
        WtT w;
        w.ldw = (N + 63) & ~63;
        w.kp  = (K + 31) & ~31;
        w.p   = W16 + wo;
        wo   += (size_t)w.ldw * w.kp;
        int tot = w.ldw * w.kp;
        cvt_wt_kernel<<<nb(tot), 256, 0, stream>>>(src, w.p, N, K, w.ldw, w.kp);
        return w;
    };
    auto gemm = [&](const _Float16* A, int lda, const WtT& w, const float* bias,
                    float* Cp, int ldc, int Mg, int Ng) {
        dim3 g((Mg + 127) / 128, (Ng + 63) / 64);
        gemm_wmma_kernel<<<g, 256, 0, stream>>>(A, lda, w.p, w.ldw, bias, Cp, ldc, Mg, Ng, w.kp);
    };
    auto bnrun = [&](float* Yp, int Cc, int Mg, const float* g, const float* be,
                     _Float16* out, int ldo, int cofs, float* stat) {
        fill(stat, 0.f, 2 * Cc);
        int splits = (Mg >= 8192) ? 64 : 2;
        int rpb = (Mg + splits - 1) / splits;
        dim3 gs(nb(Cc), splits);
        bn_stats_kernel<<<gs, 256, 0, stream>>>(Yp, stat, Mg, Cc, rpb);
        bn_apply_kernel<<<nb(Mg * Cc), 256, 0, stream>>>(Yp, stat, g, be, out, Mg, Cc, ldo, cofs, 1.f / Mg);
    };

    // ---- weights -> transposed/padded f16 ----
    WtT w_ic1a = wcvt(ic1a.w, 64, 45);
    WtT w_ic1b = wcvt(ic1b.w, 64, 64);
    WtT w_iva  = wcvt(iva.w, 64, 64);
    WtT w_ivw  = wcvt(ivw, 3, 64);
    WtT w_fc[4], w_c1[4], w_gf[4];
    for (int i = 0; i < 4; ++i) {
        w_fc[i] = wcvt(fcb[i].w, 32, 64 * (i + 1));
        w_c1[i] = wcvt(c1b[i].w, 32, 32);
        w_gf[i] = wcvt(gfb[i].w, 32, 32);
    }
    WtT w_ic2 = wcvt(ic2.w, 1024, 320);
    WtT w_f1  = wcvt(f1.w, 512, 1344);
    WtT w_f2  = wcvt(f2.w, 256, 512);
    WtT w_f3  = wcvt(f3.w, 256, 256);
    WtT w_f4  = wcvt(f4.w, 128, 256);
    WtT w_fw  = wcvt(fw, 50, 128);

    // ---- value0: xyz + 7x grid_avg -> A0 (M x 64 f16, 45 used) ----
    init_act0_kernel<<<nb(Mtot), 256, 0, stream>>>(pc, A0, Mtot);
    for (int gi = 0; gi < 7; ++gi) {
        int gs = GRIDS[gi];
        int ncell = 8 * gs * gs * gs;
        fill(CELL, 0.f, ncell * 4);
        grid_scatter_kernel<<<nb(Mtot), 256, 0, stream>>>(pc, CELL, gs, Mtot, Npts);
        grid_gather_kernel<<<nb(Mtot), 256, 0, stream>>>(pc, CELL, A0, 3 + 6 * gi, gs, Mtot, Npts);
    }

    // ---- ic1a / ic1b ----
    gemm(A0, 64, w_ic1a, nullptr, Y, 64, Mtot, 64);
    bnrun(Y, 64, Mtot, ic1a.g, ic1a.be, AA, 64, 0, STAT);
    gemm(AA, 64, w_ic1b, nullptr, Y, 64, Mtot, 64);
    bnrun(Y, 64, Mtot, ic1b.g, ic1b.be, VAL, 320, 0, STAT);

    // ---- initial velocity ----
    gemm(VAL, 320, w_iva, nullptr, Y, 64, Mtot, 64);
    bnrun(Y, 64, Mtot, iva.g, iva.be, AB, 64, 0, STAT);
    gemm(AB, 64, w_ivw, ivb, VEL, 3, Mtot, 3);
    add3_kernel<<<nb(Mtot * 3), 256, 0, stream>>>(pc, VEL, P, Mtot * 3);

    // ---- T=4 advection steps ----
    for (int t = 0; t < 4; ++t) {
        int vofs = 64 * (t + 1);      // feature columns
        int c_val1 = vofs + 32;       // value1 columns
        // value1 = conv1d_bn_relu(value)
        gemm(VAL, 320, w_fc[t], nullptr, Y, 32, Mtot, 32);
        bnrun(Y, 32, Mtot, fcb[t].g, fcb[t].be, VAL, 320, c_val1, STAT);
        // splat value1 -> 6^3 grid
        fill(FNUM, 0.f, Bv * 32);
        fill(FDEN, 0.f, Bv);
        splat_scatter_kernel<<<nb(Mtot), 256, 0, stream>>>(P, VAL, nullptr, 320, c_val1, 32,
                                                           FNUM, FDEN, Mtot, Npts, G);
        field_div_f16_kernel<<<nb(Bv * 32), 256, 0, stream>>>(FNUM, FDEN, F16A, Bv, 32);
        // c1 (1x1 conv) + BN
        gemm(F16A, 32, w_c1[t], nullptr, FY, 32, Bv, 32);
        bnrun(FY, 32, Bv, c1b[t].g, c1b[t].be, F16B, 32, 0, FST);
        // c2 (3x3x3 conv) + BN -> field
        conv3d_k3_kernel<<<nb(Bv * 32), 256, 0, stream>>>(F16B, c2b[t].w, c2b[t].b, FY, 8, G, 32);
        bnrun(FY, 32, Bv, c2b[t].g, c2b[t].be, F16C, 32, 0, FST);
        // gf (1x1 conv) + BN, sampled as feature
        gemm(F16C, 32, w_gf[t], nullptr, FY, 32, Bv, 32);
        bnrun(FY, 32, Bv, gfb[t].g, gfb[t].be, F16B, 32, 0, FST);
        sample_feature_kernel<<<nb(Mtot), 256, 0, stream>>>(F16B, P, VAL, 320, vofs, Mtot, Npts, G);
        // new_vel grid + velocity splat (same weights -> reuse FDEN)
        newvel_kernel<<<nb(Bv), 256, 0, stream>>>(F16C, gvw[t], gvb[t], NVEL, Bv, 32);
        fill(VG, 0.f, Bv * 3);
        splat_scatter_kernel<<<nb(Mtot), 256, 0, stream>>>(P, nullptr, VEL, 3, 0, 3,
                                                           VG, nullptr, Mtot, Npts, G);
        field_div_f32_kernel<<<nb(Bv * 3), 256, 0, stream>>>(VG, FDEN, VG, Bv, 3);
        flip_update_kernel<<<nb(Mtot), 256, 0, stream>>>(P, VEL, NVEL, VG, Mtot, Npts, G);
    }

    // ---- ic2 + fused BN/ReLU/max -> gmax ----
    gemm(VAL, 320, w_ic2, nullptr, Y, 1024, Mtot, 1024);
    fill(STAT, 0.f, 2048);
    {
        dim3 gs(nb(1024), 64);
        bn_stats_kernel<<<gs, 256, 0, stream>>>(Y, STAT, Mtot, 1024, 1024);
        dim3 gm(nb(1024), 8);
        bn_max_kernel<<<gm, 256, 0, stream>>>(Y, STAT, ic2.g, ic2.be, GMAX, Npts, 1024, 1.f / Mtot);
    }
    // ---- h = [value | gmax] ----
    build_h_kernel<<<nb(Mtot * 1344), 256, 0, stream>>>(VAL, GMAX, H, Mtot, Npts);

    // ---- MLP head ----
    gemm(H, 1344, w_f1, nullptr, Y, 512, Mtot, 512);
    bnrun(Y, 512, Mtot, f1.g, f1.be, H, 512, 0, STAT);
    gemm(H, 512, w_f2, nullptr, Y, 256, Mtot, 256);
    bnrun(Y, 256, Mtot, f2.g, f2.be, H, 256, 0, STAT);
    gemm(H, 256, w_f3, nullptr, Y, 256, Mtot, 256);
    bnrun(Y, 256, Mtot, f3.g, f3.be, H, 256, 0, STAT);
    gemm(H, 256, w_f4, nullptr, Y, 128, Mtot, 128);
    bnrun(Y, 128, Mtot, f4.g, f4.be, H, 128, 0, STAT);
    gemm(H, 128, w_fw, fb, Y, 50, Mtot, 50);
    log_softmax_kernel<<<nb(Mtot), 256, 0, stream>>>(Y, (float*)d_out, Mtot, 50);

    // ---- penalty ----
    fill(PENS, 0.f, 256);
    pen_pass1_kernel<<<nb(Mtot), 256, 0, stream>>>(P, seg, PENS, Mtot, Npts);
    pen_pass2_kernel<<<nb(Mtot), 256, 0, stream>>>(P, seg, PENS, Mtot, Npts);
    pen_final_kernel<<<1, 1, 0, stream>>>(PENS, (float*)d_out + (size_t)Mtot * 50);
}
